// LENSWithGAT_34952443855073
// MI455X (gfx1250) — matrix operand order repaired and checked by reading, not compiled, not run on hardware
//
#include <hip/hip_runtime.h>

// ---------------------------------------------------------------------------
// Problem constants (match reference)
// ---------------------------------------------------------------------------
#define BB   16
#define NN   512
#define FF   128
#define EE   32
#define HH   8
#define D0   64     // GHID
#define HD0  512    // HEADS*GHID
#define HIDD 256
#define HD1  2048   // HEADS*HID
#define NCLS 10
#define MTOT (BB*NN)        // 8192
#define NEGINF (-1e9f)

typedef __attribute__((ext_vector_type(16))) __bf16 v16bf;
typedef __attribute__((ext_vector_type(8)))  __bf16 v8bf;
typedef __attribute__((ext_vector_type(8)))  float  v8f;

// ---------------------------------------------------------------------------
// WMMA helpers (CDNA5 wave32, 16x16x32 bf16 -> f32)
// ---------------------------------------------------------------------------
__device__ __forceinline__ v16bf combine16(v8bf lo, v8bf hi) {
    v16bf r;
#pragma unroll
    for (int i = 0; i < 8; ++i) { r[i] = lo[i]; r[i + 8] = hi[i]; }
    return r;
}

// A fragment: lane (m = lane&15, g = lane>>4) holds row m, K chunks
// [k0+g*8 .. +7] and [k0+16+g*8 .. +7]  (row-major bf16 source)
__device__ __forceinline__ v16bf load_afrag(const __bf16* base /* row + k0 + g*8 */) {
    v8bf lo = *(const v8bf*)(base);
    v8bf hi = *(const v8bf*)(base + 16);
    return combine16(lo, hi);
}

// B fragment from packed layout: 32 bytes contiguous per lane
__device__ __forceinline__ v16bf load_bfrag(const __bf16* p) {
    v8bf lo = *(const v8bf*)(p);
    v8bf hi = *(const v8bf*)(p + 8);
    return combine16(lo, hi);
}

__device__ __forceinline__ v8f wmma_bf16(v16bf a, v16bf b, v8f c) {
    return __builtin_amdgcn_wmma_f32_16x16x32_bf16(false, a, false, b, (short)0, c, false, false);
}

// ---------------------------------------------------------------------------
// Pack a row-major [K x Nc] matrix (f32 or bf16 source) into WMMA B layout:
// tile (tk, tn): 32 lanes x 16 bf16; lane(g = lane>>4, n = lane&15),
// element e -> src[(tk*32 + (e>>3)*16 + g*8 + (e&7))*ld + tn*16 + n]
// ---------------------------------------------------------------------------
template<bool SRCBF>
__global__ __launch_bounds__(32)
void pack_b_kernel(const float* __restrict__ srcF, const __bf16* __restrict__ srcB,
                   int ld, int Nc, __bf16* __restrict__ dst,
                   long srcBatchStride, long dstBatchStride)
{
    const int lane = threadIdx.x;
    const int g = lane >> 4, n = lane & 15;
    const int tnStride = Nc >> 4;
    const int tk = blockIdx.x / tnStride;
    const int tn = blockIdx.x % tnStride;
    const int bt = blockIdx.y;

    __bf16 tmp[16];
#pragma unroll
    for (int e = 0; e < 16; ++e) {
        long sidx = (long)bt * srcBatchStride + (long)(tk * 32 + (e >> 3) * 16 + g * 8 + (e & 7)) * ld + tn * 16 + n;
        tmp[e] = SRCBF ? srcB[sidx] : (__bf16)srcF[sidx];
    }
    __bf16* d = dst + (long)bt * dstBatchStride + ((long)blockIdx.x * 32 + lane) * 16;
#pragma unroll
    for (int e = 0; e < 16; ++e) d[e] = tmp[e];
}

// ---------------------------------------------------------------------------
// Generic bf16 GEMM: C[M,Nc] = A[M,K] @ B[K,Nc]; one wave per 16x(16*NT) tile.
// MODE: 0 = store f32 only, 1 = store bf16 only, 2 = store both
// ---------------------------------------------------------------------------
template<int NT, int MODE>
__global__ __launch_bounds__(32)
void gemm_bf16_kernel(const __bf16* __restrict__ A, int lda,
                      const __bf16* __restrict__ Bp, int K, int Nc,
                      float* __restrict__ Cf, __bf16* __restrict__ Cb, int ldc)
{
    const int lane = threadIdx.x & 31;
    const int g  = lane >> 4;
    const int ln = lane & 15;
    const int row0 = blockIdx.x * 16;
    const int col0 = blockIdx.y * 16 * NT;
    const int tnStride = Nc >> 4;

    v8f acc[NT];
#pragma unroll
    for (int t = 0; t < NT; ++t)
#pragma unroll
        for (int e = 0; e < 8; ++e) acc[t][e] = 0.f;

    const __bf16* arow = A + (long)(row0 + ln) * lda + g * 8;
    for (int k0 = 0; k0 < K; k0 += 32) {
        __builtin_prefetch(arow + k0 + 64, 0, 1);   // global_prefetch_b8
        v16bf a = load_afrag(arow + k0);
#pragma unroll
        for (int t = 0; t < NT; ++t) {
            const __bf16* bp = Bp + ((long)((k0 >> 5) * tnStride + (col0 >> 4) + t) * 32 + lane) * 16;
            v16bf b = load_bfrag(bp);
            acc[t] = wmma_bf16(a, b, acc[t]);
        }
    }
    // C layout: element e -> row = e + g*8, col = ln
#pragma unroll
    for (int t = 0; t < NT; ++t)
#pragma unroll
        for (int e = 0; e < 8; ++e) {
            long idx = (long)(row0 + e + g * 8) * ldc + col0 + t * 16 + ln;
            if constexpr (MODE != 1) Cf[idx] = acc[t][e];
            if constexpr (MODE >= 1) Cb[idx] = (__bf16)acc[t][e];
        }
}

// ---------------------------------------------------------------------------
// L2 normalize node features; emit f32 + bf16 copies. block=128 (=F), one node/block
// ---------------------------------------------------------------------------
__global__ __launch_bounds__(128)
void norm_kernel(const float* __restrict__ x, float* __restrict__ xn, __bf16* __restrict__ xnb)
{
    const int node = blockIdx.x, tid = threadIdx.x;
    __shared__ float red[128];
    float v = x[(long)node * FF + tid];
    red[tid] = v * v; __syncthreads();
    for (int s = 64; s > 0; s >>= 1) { if (tid < s) red[tid] += red[tid + s]; __syncthreads(); }
    float o = v / (sqrtf(red[0]) + 1e-12f);
    xn[(long)node * FF + tid] = o;
    xnb[(long)node * FF + tid] = (__bf16)o;
}

// ---------------------------------------------------------------------------
// Edge-scoring epilogue: emb = relu(raw + b); s_src = emb.asrc; s_dst = emb.adst
// ---------------------------------------------------------------------------
__global__ void edge_score_kernel(const float* __restrict__ emb_raw, const float* __restrict__ es_b,
                                  const float* __restrict__ asrc, const float* __restrict__ adst,
                                  float* __restrict__ s_src, float* __restrict__ s_dst)
{
    int node = blockIdx.x * blockDim.x + threadIdx.x;
    if (node >= MTOT) return;
    float as = 0.f, ad = 0.f;
#pragma unroll
    for (int e = 0; e < EE; ++e) {
        float v = fmaxf(emb_raw[(long)node * EE + e] + es_b[e], 0.f);
        as += v * asrc[e]; ad += v * adst[e];
    }
    s_src[node] = as; s_dst[node] = ad;
}

// ---------------------------------------------------------------------------
// edge_weights row: ew = sigmoid(s_i + s_j)*adj; writes d_out region, row sums
// ---------------------------------------------------------------------------
__global__ __launch_bounds__(128)
void edge_weight_kernel(const float* __restrict__ s_src, const float* __restrict__ s_dst,
                        const float* __restrict__ adj, float* __restrict__ ew,
                        float* __restrict__ imp, float* __restrict__ rowadj)
{
    const int rowg = blockIdx.x;         // b*N + i
    const int b = rowg >> 9;
    const int tid = threadIdx.x;
    __shared__ float r1[128], r2[128];
    const float ssrc = s_src[rowg];
    const float* arow = adj + (long)rowg * NN;
    float* erow = ew + (long)rowg * NN;
    float sew = 0.f, sad = 0.f;
    for (int j = tid; j < NN; j += 128) {
        float a = arow[j];
        float w = 0.f;
        if (a > 0.f) w = a * (1.f / (1.f + __expf(-(ssrc + s_dst[b * NN + j]))));
        erow[j] = a * w;                  // adjs * edge_weights
        sew += w; sad += a;
    }
    r1[tid] = sew; r2[tid] = sad; __syncthreads();
    for (int s = 64; s > 0; s >>= 1) {
        if (tid < s) { r1[tid] += r1[tid + s]; r2[tid] += r2[tid + s]; }
        __syncthreads();
    }
    if (tid == 0) { imp[rowg] = r1[0]; rowadj[rowg] = r2[0]; }
}

// ---------------------------------------------------------------------------
// Per-head attention source/dest scores: ss[n,h] = h_row[h*D..] . a_src[h]
// ---------------------------------------------------------------------------
__global__ void gat_score_f32(const float* __restrict__ h, const float* __restrict__ asrc,
                              const float* __restrict__ adst, float* __restrict__ ss,
                              float* __restrict__ sd, int HD, int D)
{
    int idx = blockIdx.x * blockDim.x + threadIdx.x;
    if (idx >= MTOT * HH) return;
    int node = idx >> 3, hd = idx & 7;
    const float* row = h + (long)node * HD + hd * D;
    const float* as = asrc + hd * D;
    const float* ad = adst + hd * D;
    float s = 0.f, d = 0.f;
    for (int k = 0; k < D; ++k) { float v = row[k]; s += v * as[k]; d += v * ad[k]; }
    ss[idx] = s; sd[idx] = d;
}

__global__ void gat_score_bf(const __bf16* __restrict__ h, const float* __restrict__ asrc,
                             const float* __restrict__ adst, float* __restrict__ ss,
                             float* __restrict__ sd, int HD, int D)
{
    int idx = blockIdx.x * blockDim.x + threadIdx.x;
    if (idx >= MTOT * HH) return;
    int node = idx >> 3, hd = idx & 7;
    const __bf16* row = h + (long)node * HD + hd * D;
    const float* as = asrc + hd * D;
    const float* ad = adst + hd * D;
    float s = 0.f, d = 0.f;
    for (int k = 0; k < D; ++k) { float v = (float)row[k]; s += v * as[k]; d += v * ad[k]; }
    ss[idx] = s; sd[idx] = d;
}

// ---------------------------------------------------------------------------
// elu -> bf16
// ---------------------------------------------------------------------------
__global__ void elu_kernel(const float* __restrict__ x, __bf16* __restrict__ y, long n)
{
    long i = (long)blockIdx.x * blockDim.x + threadIdx.x;
    if (i >= n) return;
    float v = x[i];
    y[i] = (__bf16)(v > 0.f ? v : (__expf(v) - 1.f));
}

// ---------------------------------------------------------------------------
// GAT attention layer. One 128-thread block per (16-row tile, batch).
// D: per-head out dim; CONCAT: layer0 (store per head) vs layer1 (mean heads).
// hpack: per-batch WMMA-packed h [N x HDtot] bf16.
// ---------------------------------------------------------------------------
template<int D, bool CONCAT>
__global__ __launch_bounds__(128)
void gat_attn_kernel(const float* __restrict__ ss, const float* __restrict__ sd,
                     const float* __restrict__ adj, const __bf16* __restrict__ hpack,
                     float* __restrict__ out, int HDtot)
{
    constexpr int NTILES = D / 16;
    constexpr int NT = NTILES / 4;           // per-wave N-tiles (4 waves)
    const int b   = blockIdx.y;
    const int m0  = blockIdx.x * 16;
    const int tid = threadIdx.x;
    const int wave = tid >> 5;
    const int lane = tid & 31;
    const int g  = lane >> 4;
    const int ln = lane & 15;
    const int r  = tid >> 3;                 // softmax row 0..15
    const int q  = tid & 7;                  // col group 0..7
    const int j0 = q * 64;

    __shared__ unsigned adjbits[16][16];
    __shared__ float sdv[NN];
    __shared__ float red[16][8];
    __shared__ alignas(16) __bf16 Pb[16][NN];

    // stage adjacency row-block once as a bitmask (1KB)
    {
        unsigned w0 = 0u, w1 = 0u;
        const float* arow = adj + ((long)(b * NN + m0 + r)) * NN + j0;
#pragma unroll
        for (int jj = 0; jj < 64; jj += 4) {
            float4 a4 = *(const float4*)(arow + jj);
            unsigned bits = (a4.x > 0.f ? 1u : 0u) | (a4.y > 0.f ? 2u : 0u) |
                            (a4.z > 0.f ? 4u : 0u) | (a4.w > 0.f ? 8u : 0u);
            if (jj < 32) w0 |= bits << jj; else w1 |= bits << (jj - 32);
        }
        adjbits[r][q * 2 + 0] = w0;
        adjbits[r][q * 2 + 1] = w1;
    }

    v8f acc[NT];
#pragma unroll
    for (int t = 0; t < NT; ++t)
#pragma unroll
        for (int e = 0; e < 8; ++e) acc[t][e] = 0.f;

    const long packBatch = (long)NN * HDtot;
    const int  tnStride  = HDtot >> 4;

    for (int hd = 0; hd < HH; ++hd) {
        __syncthreads();                         // Pb/adjbits/sdv reuse fence
        for (int j = tid; j < NN; j += 128) sdv[j] = sd[((long)(b * NN + j)) * HH + hd];
        const float ssr = ss[((long)(b * NN + m0 + r)) * HH + hd];
        __syncthreads();

        // pass 1: row max of masked leaky-relu scores
        float mx = -1e30f;
        for (int jj = 0; jj < 64; ++jj) {
            int j = j0 + jj;
            float v = ssr + sdv[j];
            v = v > 0.f ? v : 0.2f * v;
            bool on = (adjbits[r][j >> 5] >> (j & 31)) & 1u;
            mx = fmaxf(mx, on ? v : NEGINF);
        }
        red[r][q] = mx; __syncthreads();
        float rowmax = red[r][0];
#pragma unroll
        for (int i = 1; i < 8; ++i) rowmax = fmaxf(rowmax, red[r][i]);
        __syncthreads();

        // pass 2: row sum of exp
        float sm = 0.f;
        for (int jj = 0; jj < 64; ++jj) {
            int j = j0 + jj;
            float v = ssr + sdv[j];
            v = v > 0.f ? v : 0.2f * v;
            bool on = (adjbits[r][j >> 5] >> (j & 31)) & 1u;
            if (on) sm += __expf(v - rowmax);
        }
        red[r][q] = sm; __syncthreads();
        float rowsum = 0.f;
#pragma unroll
        for (int i = 0; i < 8; ++i) rowsum += red[r][i];
        const float inv = 1.f / rowsum;

        // pass 3: write normalized probabilities as bf16
        for (int jj = 0; jj < 64; ++jj) {
            int j = j0 + jj;
            float v = ssr + sdv[j];
            v = v > 0.f ? v : 0.2f * v;
            bool on = (adjbits[r][j >> 5] >> (j & 31)) & 1u;
            Pb[r][j] = (__bf16)(on ? __expf(v - rowmax) * inv : 0.f);
        }
        __syncthreads();

        // GEMM: out_tile[16 x D] = P[16 x N] @ h_head[N x D]
        v8f cacc[NT];
#pragma unroll
        for (int t = 0; t < NT; ++t)
#pragma unroll
            for (int e = 0; e < 8; ++e) cacc[t][e] = 0.f;

        for (int kt = 0; kt < NN / 32; ++kt) {
            const int k0 = kt * 32;
            v16bf a = load_afrag(&Pb[ln][k0 + g * 8]);   // ds_load_b128 x2
#pragma unroll
            for (int t = 0; t < NT; ++t) {
                int tn = hd * NTILES + wave * NT + t;
                const __bf16* bp = hpack + (long)b * packBatch +
                                   ((long)(kt * tnStride + tn) * 32 + lane) * 16;
                v16bf bf = load_bfrag(bp);
                cacc[t] = wmma_bf16(a, bf, cacc[t]);
            }
        }

        if constexpr (CONCAT) {
#pragma unroll
            for (int t = 0; t < NT; ++t)
#pragma unroll
                for (int e = 0; e < 8; ++e) {
                    int row = m0 + e + g * 8;
                    int col = hd * D + (wave * NT + t) * 16 + ln;
                    out[((long)b * NN + row) * (HH * D) + col] = cacc[t][e];
                }
        } else {
#pragma unroll
            for (int t = 0; t < NT; ++t) acc[t] += cacc[t];
        }
    }

    if constexpr (!CONCAT) {
#pragma unroll
        for (int t = 0; t < NT; ++t)
#pragma unroll
            for (int e = 0; e < 8; ++e) {
                int row = m0 + e + g * 8;
                int col = (wave * NT + t) * 16 + ln;
                out[((long)b * NN + row) * D + col] = acc[t][e] * 0.125f;  // mean over 8 heads
            }
    }
}

// ---------------------------------------------------------------------------
// Pooling: softmax(imp masked) then graph_rep[b,d] = sum_n pa[n]*gat_out[b,n,d]
// block = 256 (= HID), grid = B
// ---------------------------------------------------------------------------
__global__ __launch_bounds__(256)
void pool_kernel(const float* __restrict__ imp, const float* __restrict__ masks,
                 const float* __restrict__ gat_out, float* __restrict__ rep)
{
    const int b = blockIdx.x, tid = threadIdx.x;
    __shared__ float pav[NN];
    __shared__ float red[256];

    float m = -1e30f;
    for (int n = tid; n < NN; n += 256) {
        float v = masks[b * NN + n] > 0.f ? imp[b * NN + n] : NEGINF;
        pav[n] = v; m = fmaxf(m, v);
    }
    red[tid] = m; __syncthreads();
    for (int s = 128; s > 0; s >>= 1) { if (tid < s) red[tid] = fmaxf(red[tid], red[tid + s]); __syncthreads(); }
    float mx = red[0]; __syncthreads();

    float sm = 0.f;
    for (int n = tid; n < NN; n += 256) { float e = __expf(pav[n] - mx); pav[n] = e; sm += e; }
    red[tid] = sm; __syncthreads();
    for (int s = 128; s > 0; s >>= 1) { if (tid < s) red[tid] += red[tid + s]; __syncthreads(); }
    float inv = 1.f / red[0]; __syncthreads();
    for (int n = tid; n < NN; n += 256) pav[n] *= inv;
    __syncthreads();

    float accv = 0.f;
    for (int n = 0; n < NN; ++n) accv += pav[n] * gat_out[((long)b * NN + n) * HIDD + tid];
    rep[b * HIDD + tid] = accv;
}

// ---------------------------------------------------------------------------
// Classifier + loss. Single block of 256 threads. Also reduces edge_reg sums.
// Writes logits [16,10] @0, labels(float) [16] @160, total_loss @176.
// ---------------------------------------------------------------------------
__global__ __launch_bounds__(256)
void head_kernel(const float* __restrict__ rep, const float* __restrict__ W1,
                 const float* __restrict__ b1, const float* __restrict__ lng,
                 const float* __restrict__ lnb, const float* __restrict__ W2,
                 const float* __restrict__ b2, const int* __restrict__ labels,
                 const float* __restrict__ imp, const float* __restrict__ rowadj,
                 float* __restrict__ out)
{
    const int tid = threadIdx.x;
    __shared__ float red[256];
    __shared__ float z[128];
    __shared__ float lg[NCLS];
    __shared__ float s_ew, s_adj, s_loss, s_mean, s_rs;

    // regularizer sums
    float pe = 0.f, pa = 0.f;
    for (int i = tid; i < MTOT; i += 256) { pe += imp[i]; pa += rowadj[i]; }
    red[tid] = pe; __syncthreads();
    for (int s = 128; s > 0; s >>= 1) { if (tid < s) red[tid] += red[tid + s]; __syncthreads(); }
    if (tid == 0) { s_ew = red[0]; s_loss = 0.f; } __syncthreads();
    red[tid] = pa; __syncthreads();
    for (int s = 128; s > 0; s >>= 1) { if (tid < s) red[tid] += red[tid + s]; __syncthreads(); }
    if (tid == 0) s_adj = red[0];
    __syncthreads();

    for (int b = 0; b < BB; ++b) {
        if (tid < 128) {
            float a = b1[tid];
            for (int d = 0; d < HIDD; ++d) a += rep[b * HIDD + d] * W1[d * 128 + tid];
            z[tid] = fmaxf(a, 0.f);
        }
        __syncthreads();
        red[tid] = (tid < 128) ? z[tid] : 0.f; __syncthreads();
        for (int s = 128; s > 0; s >>= 1) { if (tid < s) red[tid] += red[tid + s]; __syncthreads(); }
        if (tid == 0) s_mean = red[0] / 128.f;
        __syncthreads();
        float mean = s_mean;
        { float d = (tid < 128) ? (z[tid] - mean) : 0.f; red[tid] = d * d; }
        __syncthreads();
        for (int s = 128; s > 0; s >>= 1) { if (tid < s) red[tid] += red[tid + s]; __syncthreads(); }
        if (tid == 0) s_rs = rsqrtf(red[0] / 128.f + 1e-5f);
        __syncthreads();
        float rs = s_rs;

        if (tid < NCLS) {
            float a = b2[tid];
            for (int j = 0; j < 128; ++j) {
                float zj = (z[j] - mean) * rs * lng[j] + lnb[j];
                a += zj * W2[j * NCLS + tid];
            }
            lg[tid] = a;
            out[b * NCLS + tid] = a;
        }
        __syncthreads();
        if (tid == 0) {
            float mx = lg[0];
            for (int i = 1; i < NCLS; ++i) mx = fmaxf(mx, lg[i]);
            float se = 0.f;
            for (int i = 0; i < NCLS; ++i) se += __expf(lg[i] - mx);
            float lse = mx + __logf(se);
            s_loss += lse - lg[labels[b]];
        }
        __syncthreads();
    }

    if (tid < BB) out[BB * NCLS + tid] = (float)labels[tid];
    if (tid == 0) out[BB * NCLS + BB] = s_loss / (float)BB + 0.01f * s_ew / (s_adj + 1e-8f);
}

// ---------------------------------------------------------------------------
// Host launcher
// ---------------------------------------------------------------------------
extern "C" void kernel_launch(void* const* d_in, const int* in_sizes, int n_in,
                              void* d_out, int out_size, void* d_ws, size_t ws_size,
                              hipStream_t stream)
{
    const float* node_feat = (const float*)d_in[0];
    const int*   labels    = (const int*)  d_in[1];
    const float* adjs      = (const float*)d_in[2];
    const float* masks     = (const float*)d_in[3];
    const float* es_W      = (const float*)d_in[4];
    const float* es_b      = (const float*)d_in[5];
    const float* es_asrc   = (const float*)d_in[6];
    const float* es_adst   = (const float*)d_in[7];
    const float* g0W       = (const float*)d_in[8];
    const float* g0as      = (const float*)d_in[9];
    const float* g0ad      = (const float*)d_in[10];
    const float* g1W       = (const float*)d_in[11];
    const float* g1as      = (const float*)d_in[12];
    const float* g1ad      = (const float*)d_in[13];
    const float* cW1       = (const float*)d_in[14];
    const float* cb1       = (const float*)d_in[15];
    const float* lng       = (const float*)d_in[16];
    const float* lnb       = (const float*)d_in[17];
    const float* cW2       = (const float*)d_in[18];
    const float* cb2       = (const float*)d_in[19];

    float* out = (float*)d_out;
    float* ew_out = out + BB * NCLS + BB + 1;   // [B,N,N] region, offset 177

    char* wsb = (char*)d_ws;
    size_t off = 0;
    auto carve = [&](size_t bytes) -> void* {
        void* p = wsb + off;
        off = (off + bytes + 255) & ~(size_t)255;
        return p;
    };

    float*  xn      = (float*) carve((size_t)MTOT * FF * 4);
    __bf16* xnb     = (__bf16*)carve((size_t)MTOT * FF * 2);
    __bf16* esWp    = (__bf16*)carve((size_t)FF * EE * 2);
    __bf16* w0p     = (__bf16*)carve((size_t)FF * HD0 * 2);
    __bf16* w1p     = (__bf16*)carve((size_t)HD0 * HD1 * 2);
    float*  emb_raw = (float*) carve((size_t)MTOT * EE * 4);
    float*  s_src   = (float*) carve((size_t)MTOT * 4);
    float*  s_dst   = (float*) carve((size_t)MTOT * 4);
    float*  h0f     = (float*) carve((size_t)MTOT * HD0 * 4);
    __bf16* h0b     = (__bf16*)carve((size_t)MTOT * HD0 * 2);
    __bf16* h0p     = (__bf16*)carve((size_t)MTOT * HD0 * 2);
    float*  ss0     = (float*) carve((size_t)MTOT * HH * 4);
    float*  sd0     = (float*) carve((size_t)MTOT * HH * 4);
    float*  out0    = (float*) carve((size_t)MTOT * HD0 * 4);
    __bf16* helu    = (__bf16*)carve((size_t)MTOT * HD0 * 2);
    __bf16* h1b     = (__bf16*)carve((size_t)MTOT * HD1 * 2);
    __bf16* h1p     = (__bf16*)carve((size_t)MTOT * HD1 * 2);
    float*  ss1     = (float*) carve((size_t)MTOT * HH * 4);
    float*  sd1     = (float*) carve((size_t)MTOT * HH * 4);
    float*  gat_out = (float*) carve((size_t)MTOT * HIDD * 4);
    float*  imp     = (float*) carve((size_t)MTOT * 4);
    float*  rowadj  = (float*) carve((size_t)MTOT * 4);
    float*  rep     = (float*) carve((size_t)BB * HIDD * 4);
    (void)ws_size; (void)in_sizes; (void)n_in; (void)out_size;

    // 1. normalize (+bf16 copy)
    norm_kernel<<<MTOT, 128, 0, stream>>>(node_feat, xn, xnb);

    // 2. pack weights to WMMA B layout (f32 -> bf16)
    pack_b_kernel<false><<<dim3((FF/32)*(EE/16), 1), 32, 0, stream>>>(es_W, nullptr, EE, EE, esWp, 0, 0);
    pack_b_kernel<false><<<dim3((FF/32)*(HD0/16), 1), 32, 0, stream>>>(g0W, nullptr, HD0, HD0, w0p, 0, 0);
    pack_b_kernel<false><<<dim3((HD0/32)*(HD1/16), 1), 32, 0, stream>>>(g1W, nullptr, HD1, HD1, w1p, 0, 0);

    // 3. edge scoring: emb GEMM (WMMA) then bias/relu/dot + edge weights
    gemm_bf16_kernel<2, 0><<<dim3(MTOT/16, 1), 32, 0, stream>>>(xnb, FF, esWp, FF, EE, emb_raw, nullptr, EE);
    edge_score_kernel<<<MTOT/256, 256, 0, stream>>>(emb_raw, es_b, es_asrc, es_adst, s_src, s_dst);
    edge_weight_kernel<<<MTOT, 128, 0, stream>>>(s_src, s_dst, adjs, ew_out, imp, rowadj);

    // 4. GAT layer 0: h0 = xn @ W0 (store f32 + bf16), scores, pack, attention
    gemm_bf16_kernel<4, 2><<<dim3(MTOT/16, HD0/64), 32, 0, stream>>>(xnb, FF, w0p, FF, HD0, h0f, h0b, HD0);
    gat_score_f32<<<(MTOT*HH)/256, 256, 0, stream>>>(h0f, g0as, g0ad, ss0, sd0, HD0, D0);
    pack_b_kernel<true><<<dim3((NN/32)*(HD0/16), BB), 32, 0, stream>>>(nullptr, h0b, HD0, HD0, h0p,
                                                                       (long)NN*HD0, (long)NN*HD0);
    gat_attn_kernel<D0, true><<<dim3(NN/16, BB), 128, 0, stream>>>(ss0, sd0, adjs, h0p, out0, HD0);

    // 5. elu -> GAT layer 1: h1 = helu @ W1 (bf16), scores, pack, attention (mean heads)
    elu_kernel<<<(MTOT*HD0)/256, 256, 0, stream>>>(out0, helu, (long)MTOT*HD0);
    gemm_bf16_kernel<4, 1><<<dim3(MTOT/16, HD1/64), 32, 0, stream>>>(helu, HD0, w1p, HD0, HD1, nullptr, h1b, HD1);
    gat_score_bf<<<(MTOT*HH)/256, 256, 0, stream>>>(h1b, g1as, g1ad, ss1, sd1, HD1, HIDD);
    pack_b_kernel<true><<<dim3((NN/32)*(HD1/16), BB), 32, 0, stream>>>(nullptr, h1b, HD1, HD1, h1p,
                                                                       (long)NN*HD1, (long)NN*HD1);
    gat_attn_kernel<HIDD, false><<<dim3(NN/16, BB), 128, 0, stream>>>(ss1, sd1, adjs, h1p, gat_out, HD1);

    // 6. pooling + classifier + loss
    pool_kernel<<<BB, 256, 0, stream>>>(imp, masks, gat_out, rep);
    head_kernel<<<1, 256, 0, stream>>>(rep, cW1, cb1, lng, lnb, cW2, cb2, labels, imp, rowadj, out);
}
